// GmmExpectedKernel_1417339208257
// MI455X (gfx1250) — matrix operand order repaired and checked by reading, not compiled, not run on hardware
//
#include <hip/hip_runtime.h>

#define U_DIM 4096
#define V_DIM 4096
#define N_DIM 1024
#define M_DIM 1024
#define D_DIM 64

typedef __attribute__((ext_vector_type(16))) __bf16 v16bf;
typedef __attribute__((ext_vector_type(8)))  float  v8f;
typedef __attribute__((__vector_size__(16))) int    v4i;   // 128-bit payload type

union Frag32B {
  uint4 q[2];
  v16bf v;
};

#if __has_builtin(__builtin_amdgcn_global_load_async_to_lds_b128) && \
    __has_builtin(__builtin_amdgcn_s_wait_asynccnt)
#define HAS_ASYNC_LDS 1
#else
#define HAS_ASYNC_LDS 0
#endif

__device__ __forceinline__ unsigned short f2bf(float x) {
  unsigned int u = __builtin_bit_cast(unsigned int, x);
  u += 0x7FFFu + ((u >> 16) & 1u);   // round-to-nearest-even
  return (unsigned short)(u >> 16);
}

#if HAS_ASYNC_LDS
typedef __attribute__((address_space(1))) v4i* as1_v4i_ptr;
typedef __attribute__((address_space(3))) v4i* as3_v4i_ptr;

__device__ __forceinline__ void async_copy16(const unsigned short* g,
                                             unsigned short* l) {
  // global_load_async_to_lds_b128: no VGPR round trip, tracked by ASYNCcnt
  __builtin_amdgcn_global_load_async_to_lds_b128(
      (as1_v4i_ptr)(g), (as3_v4i_ptr)(l), 0, 0);
}
#endif

// ---------------------------------------------------------------- cvt f32->bf16
__global__ __launch_bounds__(256) void cvt_bf16(const float* __restrict__ s,
                                                unsigned short* __restrict__ d,
                                                int n) {
  int i = blockIdx.x * blockDim.x + threadIdx.x;
  if (i < n) d[i] = f2bf(s[i]);
}

// ---------------------------------------------------------------- Gaussian matrix
// G[n,m] = exp(0.5*(-sum log s - D*log(2pi) - sum (dmu^2)/s)), s = sp+sq
// Writes G (f32, row-major [N,M]) to gout and G^T (bf16, [M,N]) to gt.
__global__ __launch_bounds__(256) void gaussian_kernel(
    const float* __restrict__ p, const float* __restrict__ q,
    float* __restrict__ gout, unsigned short* __restrict__ gt) {
  __shared__ float pb[16][129];   // +1 pad: kills 16-way bank conflicts
  __shared__ float qb[16][129];
  const int tx = threadIdx.x, ty = threadIdx.y;
  const int tid = ty * 16 + tx;
  const int n0 = blockIdx.y * 16, m0 = blockIdx.x * 16;

  for (int idx = tid; idx < 16 * 128; idx += 256) {
    int r = idx >> 7, c = idx & 127;
    pb[r][c] = p[(size_t)(n0 + r) * 128 + c];
    qb[r][c] = q[(size_t)(m0 + r) * 128 + c];
  }
  __syncthreads();

  float l2 = 0.f, quad = 0.f;
#pragma unroll 8
  for (int d = 0; d < D_DIM; ++d) {
    float s  = pb[ty][64 + d] + qb[tx][64 + d];
    float dm = pb[ty][d]      - qb[tx][d];
    l2   += __builtin_amdgcn_logf(s);                    // v_log_f32 (log2)
    quad += dm * dm * __builtin_amdgcn_rcpf(s);          // v_rcp_f32
  }
  const float L2E = 1.4426950408889634f;
  const float CST = 117.64413150581332f;                 // D * ln(2*pi)
  float e2 = -0.5f * (l2 + (quad + CST) * L2E);
  float g  = __builtin_amdgcn_exp2f(e2);                 // v_exp_f32

  int n = n0 + ty, m = m0 + tx;
  __builtin_nontemporal_store(g, &gout[(size_t)n * M_DIM + m]);  // never re-read
  gt[(size_t)m * N_DIM + n] = f2bf(g);                           // re-read by GEMM1
}

// ---------------------------------------------------------------- NT bf16 WMMA GEMM
// C[i,j] = sum_k A[i,k]*B[j,k];  A:[I,K] bf16, B:[J,K] bf16 (both row stride K).
// mode==0: Cb[i*J+j] = bf16(C);  mode==1: Cf[i*J+j] = ln(C).
#define BM 128
#define BN 64
#define BK 32
#define AST 40   // padded LDS row stride (ushorts) = 80B -> conflict-free b128 gathers

__global__ __launch_bounds__(128) void gemm_nt_bf16(
    const unsigned short* __restrict__ A, const unsigned short* __restrict__ B,
    float* __restrict__ Cf, unsigned short* __restrict__ Cb,
    int K, int J, int mode) {
  __shared__ __align__(16) unsigned short As[2][BM * AST];  // 20.0 KB
  __shared__ __align__(16) unsigned short Bs[2][BN * AST];  // 10.0 KB

  const int tid  = threadIdx.x;
  const int lane = tid & 31;
  const int wid  = tid >> 5;
  const int r16  = lane & 15;
  const int hi   = lane >> 4;            // 0 | 1
  const int wrow = (wid >> 1) * 64;      // wave's 64-row slab
  const int wcol = (wid & 1) * 32;       // wave's 32-col slab
  const long i0 = (long)blockIdx.x * BM;
  const long j0 = (long)blockIdx.y * BN;

  const v8f vzero = {0.f, 0.f, 0.f, 0.f, 0.f, 0.f, 0.f, 0.f};
  v8f acc[4][2];
#pragma unroll
  for (int x = 0; x < 4; ++x)
#pragma unroll
    for (int y = 0; y < 2; ++y) acc[x][y] = vzero;

  // Stage one BK-deep K-slab of A (128x32) and B (64x32) into LDS buffer `buf`.
  auto stage = [&](int buf, int k0) {
#if HAS_ASYNC_LDS
#pragma unroll
    for (int rr = 0; rr < 4; ++rr) {
      int idx = tid + rr * 128;
      int row = idx >> 2;
      int c   = idx & 3;
      async_copy16(A + (size_t)(i0 + row) * K + k0 + c * 8,
                   &As[buf][row * AST] + c * 8);
    }
#pragma unroll
    for (int rr = 0; rr < 2; ++rr) {
      int idx = tid + rr * 128;
      int row = idx >> 2;
      int c   = idx & 3;
      async_copy16(B + (size_t)(j0 + row) * K + k0 + c * 8,
                   &Bs[buf][row * AST] + c * 8);
    }
#else
    // Fallback: batch loads, then batch stores -> one descending loadcnt chain.
    uint4 ra[4], rb[2];
#pragma unroll
    for (int rr = 0; rr < 4; ++rr) {
      int idx = tid + rr * 128, row = idx >> 2, c = idx & 3;
      ra[rr] = *(const uint4*)(A + (size_t)(i0 + row) * K + k0 + c * 8);
    }
#pragma unroll
    for (int rr = 0; rr < 2; ++rr) {
      int idx = tid + rr * 128, row = idx >> 2, c = idx & 3;
      rb[rr] = *(const uint4*)(B + (size_t)(j0 + row) * K + k0 + c * 8);
    }
#pragma unroll
    for (int rr = 0; rr < 4; ++rr) {
      int idx = tid + rr * 128, row = idx >> 2, c = idx & 3;
      *((uint4*)&As[buf][row * AST] + c) = ra[rr];
    }
#pragma unroll
    for (int rr = 0; rr < 2; ++rr) {
      int idx = tid + rr * 128, row = idx >> 2, c = idx & 3;
      *((uint4*)&Bs[buf][row * AST] + c) = rb[rr];
    }
#endif
  };

  const int nk = K / BK;
  stage(0, 0);
#if HAS_ASYNC_LDS
  __builtin_amdgcn_s_wait_asynccnt(0);
#endif
  __syncthreads();

  for (int t = 0; t < nk; ++t) {
    const int cur = t & 1;
    if (t + 1 < nk) stage(cur ^ 1, (t + 1) * BK);   // fill other buffer (async)
    if (t + 2 < nk) {                                // gfx1250 global_prefetch_b8
      __builtin_prefetch(A + (size_t)(i0 + tid) * K + (t + 2) * BK, 0, 1);
      if (tid < BN)
        __builtin_prefetch(B + (size_t)(j0 + tid) * K + (t + 2) * BK, 0, 1);
    }

    // A fragment, 16x32 bf16 (ISA 7.12.2): lane<16 -> K {0..7,16..23}; lane>=16 -> {8..15,24..31}
    Frag32B a[4], b[2];
#pragma unroll
    for (int x = 0; x < 4; ++x) {
      const unsigned short* ap = &As[cur][(wrow + x * 16 + r16) * AST];
      a[x].q[0] = *(const uint4*)(ap + hi * 8);
      a[x].q[1] = *(const uint4*)(ap + 16 + hi * 8);
    }
  // B fragment, 32x16: lane = output column j, K {0..7,16..23} / {8..15,24..31}
#pragma unroll
    for (int y = 0; y < 2; ++y) {
      const unsigned short* bp = &Bs[cur][(wcol + y * 16 + r16) * AST + hi * 8];
      b[y].q[0] = *(const uint4*)(bp);
      b[y].q[1] = *(const uint4*)(bp + 16);
    }

#pragma unroll
    for (int x = 0; x < 4; ++x)
#pragma unroll
      for (int y = 0; y < 2; ++y)
        acc[x][y] = __builtin_amdgcn_wmma_f32_16x16x32_bf16(
            false, a[x].v, false, b[y].v, (short)0, acc[x][y], false, false);

#if HAS_ASYNC_LDS
    __builtin_amdgcn_s_wait_asynccnt(0);  // next buffer resident before barrier
#endif
    __syncthreads();
  }

  // C layout: vgpr e, lane L -> row = e + 8*(L>=16), col = L&15
#pragma unroll
  for (int x = 0; x < 4; ++x) {
#pragma unroll
    for (int y = 0; y < 2; ++y) {
      const long col = j0 + wcol + y * 16 + r16;
      const long rb  = i0 + wrow + x * 16 + hi * 8;
#pragma unroll
      for (int e = 0; e < 8; ++e) {
        float v = acc[x][y][e];
        size_t off = (size_t)(rb + e) * J + col;
        if (mode)
          __builtin_nontemporal_store(
              0.6931471805599453f * __builtin_amdgcn_logf(v), &Cf[off]);  // ln
        else
          Cb[off] = f2bf(v);
      }
    }
  }
}

// ---------------------------------------------------------------- launcher
extern "C" void kernel_launch(void* const* d_in, const int* in_sizes, int n_in,
                              void* d_out, int out_size, void* d_ws, size_t ws_size,
                              hipStream_t stream) {
  (void)in_sizes; (void)n_in; (void)out_size; (void)ws_size;
  const float* k1 = (const float*)d_in[0];   // [U,N]
  const float* k2 = (const float*)d_in[1];   // [V,M]
  const float* p  = (const float*)d_in[2];   // [N,2D]
  const float* q  = (const float*)d_in[3];   // [M,2D]

  float* out  = (float*)d_out;                         // [U,V] fp32
  float* gout = out + (size_t)U_DIM * V_DIM;           // [N,M] fp32

  // workspace: k1b(8MB) | k2b(8MB) | Gt(2MB) | Tb(8MB) = 26 MB
  unsigned short* k1b = (unsigned short*)d_ws;
  unsigned short* k2b = k1b + (size_t)U_DIM * N_DIM;
  unsigned short* Gt  = k2b + (size_t)V_DIM * M_DIM;
  unsigned short* Tb  = Gt  + (size_t)M_DIM * N_DIM;

  cvt_bf16<<<(U_DIM * N_DIM + 255) / 256, 256, 0, stream>>>(k1, k1b, U_DIM * N_DIM);
  cvt_bf16<<<(V_DIM * M_DIM + 255) / 256, 256, 0, stream>>>(k2, k2b, V_DIM * M_DIM);

  gaussian_kernel<<<dim3(M_DIM / 16, N_DIM / 16), dim3(16, 16), 0, stream>>>(
      p, q, gout, Gt);

  // T = k1 @ G : A=k1b [4096,1024], B=Gt (G^T rows), K=1024, J=1024 -> Tb bf16
  gemm_nt_bf16<<<dim3(U_DIM / BM, M_DIM / BN), 128, 0, stream>>>(
      k1b, Gt, nullptr, Tb, N_DIM, M_DIM, 0);

  // out = ln(T @ k2^T) : A=Tb [4096,1024], B=k2b [4096,1024], K=1024, J=4096
  gemm_nt_bf16<<<dim3(U_DIM / BM, V_DIM / BN), 128, 0, stream>>>(
      Tb, k2b, out, nullptr, M_DIM, V_DIM, 1);
}
